// ExpertsChooseParallelBlock_56487409877317
// MI455X (gfx1250) — matrix operand order repaired and checked by reading, not compiled, not run on hardware
//
#include <hip/hip_runtime.h>

#define BATCH 8
#define SEQ   1024
#define DIMC  768
#define NEXP  4
#define NHEAD 12
#define HDIM  64
#define MLPH  3072
#define F1OUT 5376
#define F2IN  3840
#define F2OUT 1536
#define CAPT  512
#define LDSP  40   // padded LDS row stride in bf16 elements (80B, 16B aligned)

typedef __attribute__((ext_vector_type(16))) __bf16 bf16x16;
typedef __attribute__((ext_vector_type(8)))  float  f32x8;

union FragU { uint4 u[2]; bf16x16 v; };

__device__ __forceinline__ unsigned short f2bf(float f) {
  unsigned int u = __float_as_uint(f);
  u += 0x7fffu + ((u >> 16) & 1u);   // round-to-nearest-even bf16
  return (unsigned short)(u >> 16);
}

// CDNA5 async global->LDS copy (per-lane, ASYNCcnt-tracked). Generic shared
// pointers carry the LDS offset in their low 32 bits.
__device__ __forceinline__ void cp_async_b128(void* lds_dst, const void* gsrc) {
  unsigned loff = (unsigned)(uintptr_t)lds_dst;
  unsigned long long g = (unsigned long long)(uintptr_t)gsrc;
  asm volatile("global_load_async_to_lds_b128 %0, %1, off"
               :: "v"(loff), "v"(g) : "memory");
}
__device__ __forceinline__ void wait_async0() {
  asm volatile("s_wait_asynccnt 0x0" ::: "memory");
}

// ---------------- router: logits -> softmax probs, laid out [b][e][n] ----------------
__global__ __launch_bounds__(256) void k_router(const float* __restrict__ x,
                                                const float* __restrict__ wr,
                                                float* __restrict__ P) {
  int t = blockIdx.x * 256 + threadIdx.x;            // token id, B*N = 8192
  const float4* xr = (const float4*)(x + (size_t)t * DIMC);
  float a0 = 0.f, a1 = 0.f, a2 = 0.f, a3 = 0.f;
  for (int i = 0; i < DIMC / 4; ++i) {
    float4 xv = xr[i];
    float4 w0 = ((const float4*)(wr          ))[i];
    float4 w1 = ((const float4*)(wr +   DIMC))[i];
    float4 w2 = ((const float4*)(wr + 2*DIMC))[i];
    float4 w3 = ((const float4*)(wr + 3*DIMC))[i];
    a0 += xv.x*w0.x + xv.y*w0.y + xv.z*w0.z + xv.w*w0.w;
    a1 += xv.x*w1.x + xv.y*w1.y + xv.z*w1.z + xv.w*w1.w;
    a2 += xv.x*w2.x + xv.y*w2.y + xv.z*w2.z + xv.w*w2.w;
    a3 += xv.x*w3.x + xv.y*w3.y + xv.z*w3.z + xv.w*w3.w;
  }
  float mx = fmaxf(fmaxf(a0, a1), fmaxf(a2, a3));
  float e0 = __expf(a0 - mx), e1 = __expf(a1 - mx), e2 = __expf(a2 - mx), e3 = __expf(a3 - mx);
  float inv = 1.f / (e0 + e1 + e2 + e3);
  int b = t >> 10, n = t & (SEQ - 1);
  size_t o = ((size_t)b * NEXP) * SEQ + n;
  P[o] = e0 * inv; P[o + SEQ] = e1 * inv; P[o + 2*SEQ] = e2 * inv; P[o + 3*SEQ] = e3 * inv;
}

// ---------------- experts-choose top-512 per (b,e): bitonic sort, stable ties --------
__global__ __launch_bounds__(1024) void k_topk(const float* __restrict__ P,
                                               int* __restrict__ idx,
                                               float* __restrict__ gate) {
  __shared__ float sv[SEQ];
  __shared__ int   si[SEQ];
  int be = blockIdx.x, t = threadIdx.x;
  sv[t] = P[(size_t)be * SEQ + t]; si[t] = t;
  __syncthreads();
  for (int k = 2; k <= SEQ; k <<= 1) {
    for (int j = k >> 1; j > 0; j >>= 1) {
      int ix = t ^ j;
      if (ix > t) {
        float va = sv[t], vb = sv[ix];
        int   ia = si[t], ib = si[ix];
        bool agtb = (va > vb) || (va == vb && ia < ib);   // desc, lower idx first on tie
        bool desc = ((t & k) == 0);
        if (desc ? !agtb : agtb) { sv[t] = vb; sv[ix] = va; si[t] = ib; si[ix] = ia; }
      }
      __syncthreads();
    }
  }
  if (t < CAPT) { idx[be * CAPT + t] = si[t]; gate[be * CAPT + t] = sv[t]; }
}

// ---------------- LayerNorm -> bf16 --------------------------------------------------
__global__ __launch_bounds__(256) void k_ln(const float* __restrict__ x,
                                            const float* __restrict__ gamma,
                                            const float* __restrict__ beta,
                                            unsigned short* __restrict__ ybf) {
  int tok = blockIdx.x, t = threadIdx.x;
  const float* xr = x + (size_t)tok * DIMC;
  float v0 = xr[t], v1 = xr[t + 256], v2 = xr[t + 512];
  float s = v0 + v1 + v2, ss = v0*v0 + v1*v1 + v2*v2;
  for (int o = 16; o; o >>= 1) { s += __shfl_down(s, o, 32); ss += __shfl_down(ss, o, 32); }
  __shared__ float sm[8], sm2[8];
  if ((t & 31) == 0) { sm[t >> 5] = s; sm2[t >> 5] = ss; }
  __syncthreads();
  float ts = 0.f, tss = 0.f;
  #pragma unroll
  for (int i = 0; i < 8; ++i) { ts += sm[i]; tss += sm2[i]; }
  float mu  = ts * (1.f / DIMC);
  float var = tss * (1.f / DIMC) - mu * mu;
  float rs  = rsqrtf(var + 1e-6f);
  size_t o = (size_t)tok * DIMC;
  ybf[o + t      ] = f2bf((v0 - mu) * rs * gamma[t      ] + beta[t      ]);
  ybf[o + t + 256] = f2bf((v1 - mu) * rs * gamma[t + 256] + beta[t + 256]);
  ybf[o + t + 512] = f2bf((v2 - mu) * rs * gamma[t + 512] + beta[t + 512]);
}

// ---------------- f32 -> bf16 convert ------------------------------------------------
__global__ __launch_bounds__(256) void k_f2bf(const float* __restrict__ src,
                                              unsigned short* __restrict__ dst, int n) {
  int i = blockIdx.x * 256 + threadIdx.x;
  if (i < n) dst[i] = f2bf(src[i]);
}

// ---------------- fc1: per-(b,e) gathered GEMM, K truncated to m[e], atomic scatter --
// 256 thr = 8 waves (2 M x 4 N), wave tile 32x64 = 2x4 WMMA, block tile 64x256,
// double-buffered async global->LDS staging.
__global__ __launch_bounds__(256) void k_fc1(const unsigned short* __restrict__ ybf,
                                             const unsigned short* __restrict__ w1bf,
                                             const float* __restrict__ b1,
                                             const int* __restrict__ idx,
                                             float* __restrict__ htok) {
  __shared__ unsigned short As[2][64 * LDSP];
  __shared__ unsigned short Bs[2][256 * LDSP];
  __shared__ int sTok[64];
  int t = threadIdx.x, lane = t & 31, wave = t >> 5;
  int wm = wave & 1, wn = wave >> 1;             // 2 x 4 wave grid
  int be = blockIdx.z, b = be >> 2, e = be & 3;
  int K = DIMC >> (3 - e);                       // 96,192,384,768 (= in_mask width)
  int m0 = blockIdx.y * 64;
  int n0 = blockIdx.x * 256;
  if (t < 64) sTok[t] = idx[be * CAPT + m0 + t];
  __syncthreads();
  const unsigned short* aRow = ybf  + ((size_t)(b * SEQ + sTok[t >> 2])) * DIMC + (t & 3) * 8;
  const unsigned short* bRow = w1bf + (size_t)(n0 + t) * DIMC;
  unsigned short* aDst0 = &As[0][(t >> 2) * LDSP + (t & 3) * 8];
  unsigned short* aDst1 = &As[1][(t >> 2) * LDSP + (t & 3) * 8];
  unsigned short* bDst0 = &Bs[0][t * LDSP];
  unsigned short* bDst1 = &Bs[1][t * LDSP];
  f32x8 acc[2][4] = {};
  int hi = lane >> 4, ml = lane & 15;

  // prologue: stage k0 = 0 into buffer 0
  cp_async_b128(aDst0, aRow);
  #pragma unroll
  for (int c = 0; c < 4; ++c) cp_async_b128(bDst0 + c * 8, bRow + c * 8);

  int buf = 0;
  for (int k0 = 0; k0 < K; k0 += 32) {
    wait_async0();             // current buffer's DMA complete (this wave)
    __syncthreads();           // all waves' DMA complete; prev reads of other buf done
    if (k0 + 32 < K) {         // overlap: stage next tile while computing this one
      unsigned short* aD = buf ? aDst0 : aDst1;
      unsigned short* bD = buf ? bDst0 : bDst1;
      cp_async_b128(aD, aRow + k0 + 32);
      #pragma unroll
      for (int c = 0; c < 4; ++c) cp_async_b128(bD + c * 8, bRow + k0 + 32 + c * 8);
    }
    FragU afr[2], bfr[4];
    #pragma unroll
    for (int tm = 0; tm < 2; ++tm) {
      int row = wm * 32 + tm * 16 + ml;
      afr[tm].u[0] = *(const uint4*)&As[buf][row * LDSP + hi * 8];
      afr[tm].u[1] = *(const uint4*)&As[buf][row * LDSP + 16 + hi * 8];
    }
    #pragma unroll
    for (int tn = 0; tn < 4; ++tn) {
      int row = wn * 64 + tn * 16 + ml;
      bfr[tn].u[0] = *(const uint4*)&Bs[buf][row * LDSP + hi * 8];
      bfr[tn].u[1] = *(const uint4*)&Bs[buf][row * LDSP + 16 + hi * 8];
    }
    #pragma unroll
    for (int tm = 0; tm < 2; ++tm)
      #pragma unroll
      for (int tn = 0; tn < 4; ++tn)
        acc[tm][tn] = __builtin_amdgcn_wmma_f32_16x16x32_bf16(
            false, afr[tm].v, false, bfr[tn].v, (short)0, acc[tm][tn], false, false);
    __syncthreads();           // done reading buf -> safe target for next-next issue
    buf ^= 1;
  }
  #pragma unroll
  for (int tm = 0; tm < 2; ++tm) {
    int cl = wm * 32 + tm * 16;
    #pragma unroll
    for (int tn = 0; tn < 4; ++tn) {
      int f0 = n0 + wn * 64 + tn * 16 + ml;
      float bias = b1[f0];
      #pragma unroll
      for (int r = 0; r < 8; ++r) {
        int tok = sTok[cl + hi * 8 + r];
        atomicAdd(&htok[((size_t)(b * SEQ + tok)) * F1OUT + f0], acc[tm][tn][r] + bias);
      }
    }
  }
}

// ---------------- attention: block = (b,h, 32 queries), flash-style online softmax ---
__global__ __launch_bounds__(256) void k_attn(const float* __restrict__ htok,
                                              float* __restrict__ xattn) {
  __shared__ float Ks[64][68];
  __shared__ float Vs[64][68];
  __shared__ float Qs[32][64];
  __shared__ float Ps[8][4][32];
  int t = threadIdx.x, lane = t & 31, w = t >> 5;
  int qb = blockIdx.x & 31;
  int bh = blockIdx.x >> 5;
  int h = bh % NHEAD, b = bh / NHEAD;
  const float* base = htok + (size_t)b * SEQ * F1OUT + MLPH + h * HDIM;
  int q0 = qb * 32;
  { int r = t >> 3, dc = (t & 7) * 8;
    const float* qp = base + (size_t)(q0 + r) * F1OUT + dc;
    *(float4*)&Qs[r][dc]     = *(const float4*)qp;
    *(float4*)&Qs[r][dc + 4] = *(const float4*)(qp + 4);
  }
  float m[4], l[4], o0[4], o1[4];
  #pragma unroll
  for (int j = 0; j < 4; ++j) { m[j] = -3.0e38f; l[j] = 0.f; o0[j] = 0.f; o1[j] = 0.f; }
  int myq = w * 4;
  for (int kc = 0; kc < SEQ; kc += 64) {
    __syncthreads();
    { int r = t >> 2, dc = (t & 3) * 16;
      const float* kp = base + (size_t)(kc + r) * F1OUT + DIMC + dc;
      const float* vp = base + (size_t)(kc + r) * F1OUT + 2 * DIMC + dc;
      #pragma unroll
      for (int i = 0; i < 4; ++i) {
        *(float4*)&Ks[r][dc + 4 * i] = *(const float4*)(kp + 4 * i);
        *(float4*)&Vs[r][dc + 4 * i] = *(const float4*)(vp + 4 * i);
      }
    }
    __syncthreads();
    #pragma unroll
    for (int sub = 0; sub < 2; ++sub) {
      int key = sub * 32 + lane;
      float s[4] = {0.f, 0.f, 0.f, 0.f};
      for (int d = 0; d < HDIM; ++d) {
        float kv = Ks[key][d];
        s[0] += Qs[myq + 0][d] * kv;
        s[1] += Qs[myq + 1][d] * kv;
        s[2] += Qs[myq + 2][d] * kv;
        s[3] += Qs[myq + 3][d] * kv;
      }
      #pragma unroll
      for (int j = 0; j < 4; ++j) {
        float sv = s[j] * 0.125f;            // HEAD_DIM^-0.5
        float cm = sv;
        for (int o = 16; o; o >>= 1) cm = fmaxf(cm, __shfl_xor(cm, o, 32));
        float mn = fmaxf(m[j], cm);
        float corr = __expf(m[j] - mn);
        float p = __expf(sv - mn);
        float ps = p;
        for (int o = 16; o; o >>= 1) ps += __shfl_xor(ps, o, 32);
        l[j] = l[j] * corr + ps;
        o0[j] *= corr; o1[j] *= corr;
        m[j] = mn;
        Ps[w][j][lane] = p;
      }
      __builtin_amdgcn_wave_barrier();
      for (int k = 0; k < 32; ++k) {
        float v0 = Vs[sub * 32 + k][lane];
        float v1 = Vs[sub * 32 + k][lane + 32];
        #pragma unroll
        for (int j = 0; j < 4; ++j) { float p = Ps[w][j][k]; o0[j] += p * v0; o1[j] += p * v1; }
      }
      __builtin_amdgcn_wave_barrier();
    }
  }
  #pragma unroll
  for (int j = 0; j < 4; ++j) {
    float inv = 1.f / l[j];
    size_t o = ((size_t)(b * SEQ + q0 + myq + j)) * DIMC + h * HDIM;
    xattn[o + lane]      = o0[j] * inv;
    xattn[o + 32 + lane] = o1[j] * inv;
  }
}

// ---------------- gelu(x_mlp) || x_attn -> bf16 --------------------------------------
__global__ __launch_bounds__(256) void k_gc(const float* __restrict__ htok,
                                            const float* __restrict__ xattn,
                                            unsigned short* __restrict__ y2bf) {
  size_t i = (size_t)blockIdx.x * 256 + threadIdx.x;   // total = B*SEQ*F2IN, exact
  size_t tokn = i / F2IN;
  int f = (int)(i % F2IN);
  float v;
  if (f < MLPH) {
    float u = htok[tokn * F1OUT + f];
    v = 0.5f * u * (1.f + erff(u * 0.70710678118f));   // exact GELU
  } else {
    v = xattn[tokn * DIMC + (f - MLPH)];
  }
  y2bf[i] = f2bf(v);
}

// ---------------- fc2: gathered GEMM, out-mask + gate, atomic residual add -----------
__global__ __launch_bounds__(256) void k_fc2(const unsigned short* __restrict__ y2bf,
                                             const unsigned short* __restrict__ w2bf,
                                             const float* __restrict__ b2,
                                             const int* __restrict__ idx,
                                             const float* __restrict__ gate,
                                             float* __restrict__ out) {
  __shared__ unsigned short As[2][64 * LDSP];
  __shared__ unsigned short Bs[2][256 * LDSP];
  __shared__ int   sTok[64];
  __shared__ float sGate[64];
  int t = threadIdx.x, lane = t & 31, wave = t >> 5;
  int wm = wave & 1, wn = wave >> 1;
  int be = blockIdx.z, b = be >> 2, e = be & 3;
  int me = DIMC >> (3 - e);
  int m0 = blockIdx.y * 64;
  int n0 = blockIdx.x * 256;
  if (t < 64) { sTok[t] = idx[be * CAPT + m0 + t]; sGate[t] = gate[be * CAPT + m0 + t]; }
  __syncthreads();
  const unsigned short* aRow = y2bf + ((size_t)(b * SEQ + sTok[t >> 2])) * F2IN + (t & 3) * 8;
  const unsigned short* bRow = w2bf + (size_t)(n0 + t) * F2IN;
  unsigned short* aDst0 = &As[0][(t >> 2) * LDSP + (t & 3) * 8];
  unsigned short* aDst1 = &As[1][(t >> 2) * LDSP + (t & 3) * 8];
  unsigned short* bDst0 = &Bs[0][t * LDSP];
  unsigned short* bDst1 = &Bs[1][t * LDSP];
  f32x8 acc[2][4] = {};
  int hi = lane >> 4, ml = lane & 15;

  cp_async_b128(aDst0, aRow);
  #pragma unroll
  for (int c = 0; c < 4; ++c) cp_async_b128(bDst0 + c * 8, bRow + c * 8);

  int buf = 0;
  for (int k0 = 0; k0 < F2IN; k0 += 32) {
    wait_async0();
    __syncthreads();
    if (k0 + 32 < F2IN) {
      unsigned short* aD = buf ? aDst0 : aDst1;
      unsigned short* bD = buf ? bDst0 : bDst1;
      cp_async_b128(aD, aRow + k0 + 32);
      #pragma unroll
      for (int c = 0; c < 4; ++c) cp_async_b128(bD + c * 8, bRow + k0 + 32 + c * 8);
    }
    FragU afr[2], bfr[4];
    #pragma unroll
    for (int tm = 0; tm < 2; ++tm) {
      int row = wm * 32 + tm * 16 + ml;
      afr[tm].u[0] = *(const uint4*)&As[buf][row * LDSP + hi * 8];
      afr[tm].u[1] = *(const uint4*)&As[buf][row * LDSP + 16 + hi * 8];
    }
    #pragma unroll
    for (int tn = 0; tn < 4; ++tn) {
      int row = wn * 64 + tn * 16 + ml;
      bfr[tn].u[0] = *(const uint4*)&Bs[buf][row * LDSP + hi * 8];
      bfr[tn].u[1] = *(const uint4*)&Bs[buf][row * LDSP + 16 + hi * 8];
    }
    #pragma unroll
    for (int tm = 0; tm < 2; ++tm)
      #pragma unroll
      for (int tn = 0; tn < 4; ++tn)
        acc[tm][tn] = __builtin_amdgcn_wmma_f32_16x16x32_bf16(
            false, afr[tm].v, false, bfr[tn].v, (short)0, acc[tm][tn], false, false);
    __syncthreads();
    buf ^= 1;
  }
  #pragma unroll
  for (int tm = 0; tm < 2; ++tm) {
    int cl = wm * 32 + tm * 16;
    #pragma unroll
    for (int tn = 0; tn < 4; ++tn) {
      int f0 = n0 + wn * 64 + tn * 16 + ml;
      int oc = (f0 < DIMC) ? f0 : f0 - DIMC;       // both halves fold onto residual
      if (oc < me) {                               // out_mask
        float bias = b2[f0];
        #pragma unroll
        for (int r = 0; r < 8; ++r) {
          int rr = cl + hi * 8 + r;
          int tok = sTok[rr];
          atomicAdd(&out[((size_t)(b * SEQ + tok)) * DIMC + oc],
                    sGate[rr] * (acc[tm][tn][r] + bias));
        }
      }
    }
  }
}

// ====================================================================================
extern "C" void kernel_launch(void* const* d_in, const int* in_sizes, int n_in,
                              void* d_out, int out_size, void* d_ws, size_t ws_size,
                              hipStream_t stream) {
  (void)in_sizes; (void)n_in; (void)out_size; (void)ws_size;
  const float* x     = (const float*)d_in[0];
  const float* wr    = (const float*)d_in[1];
  const float* gamma = (const float*)d_in[2];
  const float* beta  = (const float*)d_in[3];
  const float* w1    = (const float*)d_in[4];
  const float* b1    = (const float*)d_in[5];
  const float* w2    = (const float*)d_in[6];
  const float* b2    = (const float*)d_in[7];
  float* out = (float*)d_out;

  char* ws = (char*)d_ws;
  size_t off = 0;
  auto carve = [&](size_t bytes) -> void* {
    void* p = (void*)(ws + off);
    off += (bytes + 255) & ~(size_t)255;
    return p;
  };
  float*          P     = (float*)         carve((size_t)BATCH * NEXP * SEQ * 4);
  int*            idx   = (int*)           carve((size_t)BATCH * NEXP * CAPT * 4);
  float*          gate  = (float*)         carve((size_t)BATCH * NEXP * CAPT * 4);
  unsigned short* ybf   = (unsigned short*)carve((size_t)BATCH * SEQ * DIMC * 2);
  unsigned short* w1bf  = (unsigned short*)carve((size_t)F1OUT * DIMC * 2);
  unsigned short* w2bf  = (unsigned short*)carve((size_t)F2OUT * F2IN * 2);
  float*          htok  = (float*)         carve((size_t)BATCH * SEQ * F1OUT * 4);
  float*          xattn = (float*)         carve((size_t)BATCH * SEQ * DIMC * 4);
  unsigned short* y2bf  = (unsigned short*)carve((size_t)BATCH * SEQ * F2IN * 2);

  // 1. router probs
  k_router<<<(BATCH * SEQ) / 256, 256, 0, stream>>>(x, wr, P);
  // 2. experts-choose top-512
  k_topk<<<BATCH * NEXP, 1024, 0, stream>>>(P, idx, gate);
  // 3. LayerNorm -> bf16
  k_ln<<<BATCH * SEQ, 256, 0, stream>>>(x, gamma, beta, ybf);
  // 4. weights -> bf16
  { int n1 = F1OUT * DIMC; k_f2bf<<<(n1 + 255) / 256, 256, 0, stream>>>(w1, w1bf, n1); }
  { int n2 = F2OUT * F2IN; k_f2bf<<<(n2 + 255) / 256, 256, 0, stream>>>(w2, w2bf, n2); }
  // 5. fc1 scatter target starts at zero
  hipMemsetAsync(htok, 0, (size_t)BATCH * SEQ * F1OUT * 4, stream);
  // 6. fc1 WMMA GEMM + scatter (K truncated to m[e] == in_mask)
  k_fc1<<<dim3(F1OUT / 256, CAPT / 64, BATCH * NEXP), 256, 0, stream>>>(ybf, w1bf, b1, idx, htok);
  // 7. attention on qkv slice of htok
  k_attn<<<BATCH * NHEAD * (SEQ / 32), 256, 0, stream>>>(htok, xattn);
  // 8. gelu(x_mlp) || x_attn -> bf16
  k_gc<<<(int)(((size_t)BATCH * SEQ * F2IN) / 256), 256, 0, stream>>>(htok, xattn, y2bf);
  // 9. residual: out = x
  hipMemcpyAsync(out, x, (size_t)BATCH * SEQ * DIMC * 4, hipMemcpyDeviceToDevice, stream);
  // 10. fc2 WMMA GEMM + masked, gated scatter into out
  k_fc2<<<dim3(F2OUT / 256, CAPT / 64, BATCH * NEXP), 256, 0, stream>>>(y2bf, w2bf, b2, idx, gate, out);
}
